// MLN_GCN_31585189495115
// MI455X (gfx1250) — matrix operand order repaired and checked by reading, not compiled, not run on hardware
//
#include <hip/hip_runtime.h>
#include <hip/hip_bf16.h>
#include <math.h>

typedef _Float16 v16h __attribute__((ext_vector_type(16)));
typedef _Float16 v8h  __attribute__((ext_vector_type(8)));
typedef _Float16 v2h  __attribute__((ext_vector_type(2)));
typedef float    v8f  __attribute__((ext_vector_type(8)));
typedef float    v4f  __attribute__((ext_vector_type(4)));

#define NUMN     512
#define DIM      64
#define EDG      4096
#define MAINNUM  10
#define LDA      72          // padded f16 row stride (bank-conflict-free fragment loads)
#define NTHREADS 512
#define NWAVES   (NTHREADS / 32)

// -------- shared memory layout (bytes) --------
#define OFF_EROW 0                         // int  [4096]  16384
#define OFF_ECOL 16384                     // int  [4096]  16384
#define OFF_NORM 32768                     // f32  [4096]  16384
#define OFF_W1T  49152                     // f16  [64*72]  9216 (transposed: [n][k])
#define OFF_W2T  58368                     // f16  [64*72]  9216
#define OFF_B1   67584                     // f32  [64]
#define OFF_B2   67840                     // f32  [64]
#define OFF_W3   68096                     // f32  [64]
#define OFF_HA   68352                     // f16  [512*72] 73728 (activations, in-place GEMM)
#define OFF_HACC 142080                    // f16  [512*64] 65536 (pk_add_f16 accumulator)
#define SMEM_BYTES (OFF_HACC + NUMN*DIM*2) // 207616 < 320KB per WGP

// -------------------------------------------------------------------------
// setup: deg -> dinv -> norm[e] = dinv[row]*dinv[col]   (one block)
// -------------------------------------------------------------------------
__global__ __launch_bounds__(512)
void gcn_setup_norm(const int* __restrict__ erow, const int* __restrict__ ecol,
                    float* __restrict__ norm) {
    __shared__ float deg[NUMN];
    __shared__ float dinv[NUMN];
    const int t = threadIdx.x;
    if (t < NUMN) deg[t] = 0.0f;
    __syncthreads();
    for (int e = t; e < EDG; e += 512) atomicAdd(&deg[ecol[e]], 1.0f);
    __syncthreads();
    if (t < NUMN) dinv[t] = (deg[t] > 0.0f) ? rsqrtf(deg[t]) : 0.0f;
    __syncthreads();
    for (int e = t; e < EDG; e += 512) norm[e] = dinv[erow[e]] * dinv[ecol[e]];
}

// -------------------------------------------------------------------------
// CDNA5 helpers: async global->LDS copy, packed f16 LDS atomic
// -------------------------------------------------------------------------
__device__ __forceinline__ void async_copy_b128(uint32_t lds_off, const void* g) {
#if defined(__gfx1250__)
    asm volatile("global_load_async_to_lds_b128 %0, %1, off"
                 :: "v"(lds_off), "v"((unsigned long long)(uintptr_t)g)
                 : "memory");
#else
    (void)lds_off; (void)g;
#endif
}

__device__ __forceinline__ void lds_pk_add_f16(uint32_t lds_off, v2h val) {
#if defined(__gfx1250__)
    union { v2h h; uint32_t u; } p; p.h = val;
    asm volatile("ds_pk_add_f16 %0, %1" :: "v"(lds_off), "v"(p.u) : "memory");
#else
    (void)lds_off; (void)val;
#endif
}

// -------------------------------------------------------------------------
// fragment helper (wave32 WMMA 16x16x32 f16 layouts per CDNA5 ISA tables)
// -------------------------------------------------------------------------
__device__ __forceinline__ v16h ld_frag16(const _Float16* p0, const _Float16* p1) {
    v8h lo = *(const v8h*)p0;   // ds_load_b128
    v8h hi = *(const v8h*)p1;   // ds_load_b128
    v16h r;
#pragma unroll
    for (int i = 0; i < 8; ++i) { r[i] = lo[i]; r[i + 8] = hi[i]; }
    return r;
}

// -------------------------------------------------------------------------
// one GCN layer: zero acc -> WMMA GEMM (in-place hA) -> edge scatter -> bias+ReLU
// -------------------------------------------------------------------------
__device__ __forceinline__ void gcn_layer(_Float16* hA, _Float16* haccH,
                                          const _Float16* Wt, const float* bias,
                                          const int* sRow, const int* sCol,
                                          const float* sNorm, int tid) {
    const int wave = tid >> 5;
    const int lane = tid & 31;
    const int lm   = lane & 15;   // M (A/D rows) or N (B/D cols) within tile
    const int g    = lane >> 4;   // half-wave group

    // zero f16 accumulator (vectorized: 64KB / 16B = 4096 b128 stores)
    for (int i = tid; i < (NUMN * DIM) / 8; i += NTHREADS)
        ((v4f*)haccH)[i] = (v4f){0.f, 0.f, 0.f, 0.f};
    __syncthreads();

    // GEMM: hW(512x64) = hA(512x64,f16) * W(64x64,f16), f32 accum, in-place
    for (int mt = wave; mt < NUMN / 16; mt += NWAVES) {
        const _Float16* arow = hA + (mt * 16 + lm) * LDA;
        // A halves 0..7 : K = kh*32 + 8g + h ; halves 8..15 : K = kh*32 + 16 + 8g + h
        v16h a0 = ld_frag16(arow + 8 * g,       arow + 16 + 8 * g);        // K 0..31
        v16h a1 = ld_frag16(arow + 32 + 8 * g,  arow + 48 + 8 * g);        // K 32..63
#pragma unroll
        for (int nt = 0; nt < 4; ++nt) {
            const _Float16* wrow = Wt + (nt * 16 + lm) * LDA;              // Wt is [n][k]
            // B halves 0..7 : K = kh*32 + 8g + h ; halves 8..15 : K = kh*32 + 16 + 8g + h   (lane = column n)
            v16h bf0 = ld_frag16(wrow + 8 * g,        wrow + 16 + 8 * g);
            v16h bf1 = ld_frag16(wrow + 32 + 8 * g,   wrow + 48 + 8 * g);
            v8f c = {0.f, 0.f, 0.f, 0.f, 0.f, 0.f, 0.f, 0.f};
            c = __builtin_amdgcn_wmma_f32_16x16x32_f16(false, a0, false, bf0,
                                                       (short)0, c, false, false);
            c = __builtin_amdgcn_wmma_f32_16x16x32_f16(false, a1, false, bf1,
                                                       (short)0, c, false, false);
            // D: VGPR r -> row mt*16 + r + 8g, col nt*16 + lm ; write back f16
#pragma unroll
            for (int r = 0; r < 8; ++r)
                hA[(mt * 16 + r + 8 * g) * LDA + nt * 16 + lm] = (_Float16)c[r];
        }
    }
    __syncthreads();

    // scatter: hacc[col, f0:f0+2] += hA[row, f0:f0+2] * norm[e]
    // one ds_load_b32 + one ds_pk_add_f16 per edge-lane
    const int f0 = lane * 2;
    const uint32_t haccBase = (uint32_t)(uintptr_t)haccH;   // DS addr = low 32 bits
    for (int e = wave; e < EDG; e += NWAVES) {
        const int      r  = sRow[e];
        const int      c  = sCol[e];
        const _Float16 nh = (_Float16)sNorm[e];
        v2h hv  = *(const v2h*)(hA + r * LDA + f0);
        v2h nm2 = {nh, nh};
        lds_pk_add_f16(haccBase + (uint32_t)(c * DIM + f0) * 2u, hv * nm2);
    }
#if defined(__gfx1250__)
    asm volatile("s_wait_dscnt 0" ::: "memory");   // asm atomics untracked by compiler
#endif
    __syncthreads();

    // hA = relu(hacc + bias) as f16
    for (int i = tid; i < NUMN * DIM; i += NTHREADS) {
        const int n = i >> 6, f = i & 63;
        float v = (float)haccH[i] + bias[f];
        hA[n * LDA + f] = (_Float16)(v > 0.f ? v : 0.f);
    }
    __syncthreads();
}

// -------------------------------------------------------------------------
// fused GCN: one workgroup per batch graph, everything resident in LDS
// -------------------------------------------------------------------------
__global__ __launch_bounds__(NTHREADS)
void gcn_fused(const float* __restrict__ x,   const float* __restrict__ emb,
               const float* __restrict__ W1,  const float* __restrict__ b1,
               const float* __restrict__ W2,  const float* __restrict__ b2,
               const float* __restrict__ W3,  const float* __restrict__ b3,
               const int* __restrict__ erow,  const int* __restrict__ ecol,
               const float* __restrict__ norm, float* __restrict__ out) {
    extern __shared__ char smem[];
    int*      sRow  = (int*)(smem + OFF_EROW);
    int*      sCol  = (int*)(smem + OFF_ECOL);
    float*    sNorm = (float*)(smem + OFF_NORM);
    _Float16* sW1t  = (_Float16*)(smem + OFF_W1T);
    _Float16* sW2t  = (_Float16*)(smem + OFF_W2T);
    float*    sB1   = (float*)(smem + OFF_B1);
    float*    sB2   = (float*)(smem + OFF_B2);
    float*    sW3   = (float*)(smem + OFF_W3);
    _Float16* hA    = (_Float16*)(smem + OFF_HA);
    _Float16* haccH = (_Float16*)(smem + OFF_HACC);

    const int tid = threadIdx.x;
    const int b   = blockIdx.x;

    // stage graph structure + norms with async global->LDS DMA (ASYNCcnt path)
#if defined(__gfx1250__)
    {
        const uint32_t lRow  = (uint32_t)(uintptr_t)sRow;
        const uint32_t lCol  = (uint32_t)(uintptr_t)sCol;
        const uint32_t lNorm = (uint32_t)(uintptr_t)sNorm;
        for (int i = tid; i < EDG / 4; i += NTHREADS) {    // b128 = 4 elems
            async_copy_b128(lRow  + i * 16u, erow + i * 4);
            async_copy_b128(lCol  + i * 16u, ecol + i * 4);
            async_copy_b128(lNorm + i * 16u, norm + i * 4);
        }
        asm volatile("s_wait_asynccnt 0" ::: "memory");
    }
#else
    for (int e = tid; e < EDG; e += NTHREADS) {
        sRow[e] = erow[e]; sCol[e] = ecol[e]; sNorm[e] = norm[e];
    }
#endif
    // stage weights transposed ([n][k]) as f16 with padded stride
    for (int i = tid; i < DIM * DIM; i += NTHREADS) {
        const int k = i >> 6, n = i & 63;
        sW1t[n * LDA + k] = (_Float16)W1[k * DIM + n];
        sW2t[n * LDA + k] = (_Float16)W2[k * DIM + n];
    }
    if (tid < DIM) { sB1[tid] = b1[tid]; sB2[tid] = b2[tid]; sW3[tid] = W3[tid]; }
    // h0[n,f] = x[b,n] * emb[n,f]
    for (int i = tid; i < NUMN * DIM; i += NTHREADS) {
        const int n = i >> 6;
        hA[n * LDA + (i & 63)] = (_Float16)(x[b * NUMN + n] * emb[i]);
    }
    __syncthreads();

    gcn_layer(hA, haccH, sW1t, sB1, sRow, sCol, sNorm, tid);
    gcn_layer(hA, haccH, sW2t, sB2, sRow, sCol, sNorm, tid);

    // layer 3: hw3[n] = hA[n,:].W3 ; agg3[col] += hw3[row]*norm ; logits = agg3 + b3
    float* hw3  = (float*)haccH;          // [512] f32, reuse accumulator region
    float* agg3 = (float*)haccH + NUMN;   // [512] f32
    for (int n = tid; n < NUMN; n += NTHREADS) {
        float s = 0.f;
#pragma unroll
        for (int f = 0; f < DIM; ++f) s += (float)hA[n * LDA + f] * sW3[f];
        hw3[n]  = s;
        agg3[n] = 0.f;
    }
    __syncthreads();
    for (int e = tid; e < EDG; e += NTHREADS)
        atomicAdd(agg3 + sCol[e], hw3[sRow[e]] * sNorm[e]);
    __syncthreads();

    const float b3v = b3[0];
    float* outb = out + (size_t)b * NUMN;
    if (tid == 0) {                      // softmax over logits[0:10]
        float mx = -1e30f;
        for (int i = 0; i < MAINNUM; ++i) {
            const float l = agg3[i] + b3v;
            mx = fmaxf(mx, l);
        }
        float ex[MAINNUM];
        float s = 0.f;
        for (int i = 0; i < MAINNUM; ++i) { ex[i] = expf(agg3[i] + b3v - mx); s += ex[i]; }
        const float inv = 1.f / s;
        for (int i = 0; i < MAINNUM; ++i) outb[i] = ex[i] * inv;
    }
    for (int n = MAINNUM + tid; n < NUMN; n += NTHREADS) {   // sigmoid tail
        const float l = agg3[n] + b3v;
        outb[n] = 1.f / (1.f + expf(-l));
    }
}

// -------------------------------------------------------------------------
extern "C" void kernel_launch(void* const* d_in, const int* in_sizes, int n_in,
                              void* d_out, int out_size, void* d_ws, size_t ws_size,
                              hipStream_t stream) {
    (void)in_sizes; (void)n_in; (void)out_size; (void)ws_size;
    const float* x    = (const float*)d_in[0];
    const float* emb  = (const float*)d_in[1];
    const float* W1   = (const float*)d_in[2];
    const float* b1   = (const float*)d_in[3];
    const float* W2   = (const float*)d_in[4];
    const float* b2   = (const float*)d_in[5];
    const float* W3   = (const float*)d_in[6];
    const float* b3   = (const float*)d_in[7];
    const int*   erow = (const int*)d_in[8];
    const int*   ecol = (const int*)d_in[9];
    float*       norm = (float*)d_ws;               // [4096] scratch
    float*       out  = (float*)d_out;              // [1024*512]

    (void)hipFuncSetAttribute((const void*)gcn_fused,
                              hipFuncAttributeMaxDynamicSharedMemorySize,
                              SMEM_BYTES);

    gcn_setup_norm<<<1, 512, 0, stream>>>(erow, ecol, norm);
    gcn_fused<<<1024, NTHREADS, SMEM_BYTES, stream>>>(x, emb, W1, b1, W2, b2,
                                                      W3, b3, erow, ecol, norm, out);
}
